// LSTMNetwork_71975061946848
// MI455X (gfx1250) — matrix operand order repaired and checked by reading, not compiled, not run on hardware
//
#include <hip/hip_runtime.h>
#include <math.h>

#define L_LAYERS 4
#define BATCH    4096
#define DIM      2048

typedef __attribute__((ext_vector_type(16))) __bf16 v16bf;
typedef __attribute__((ext_vector_type(8)))  float  v8f;
typedef __attribute__((ext_vector_type(4)))  unsigned int v4u;
typedef int v4i_gcc __attribute__((vector_size(16)));   // matches builtin param type

__device__ __forceinline__ unsigned short f2bf(float f) {
  unsigned int u = __builtin_bit_cast(unsigned int, f);
  u += 0x7FFFu + ((u >> 16) & 1u);               // round-to-nearest-even
  return (unsigned short)(u >> 16);
}
__device__ __forceinline__ float fast_rcp(float x) {
#if __has_builtin(__builtin_amdgcn_rcpf)
  return __builtin_amdgcn_rcpf(x);
#else
  return 1.0f / x;
#endif
}
__device__ __forceinline__ float fast_sig(float x) { return fast_rcp(1.0f + __expf(-x)); }
__device__ __forceinline__ float fast_tanh(float x) {
#if __has_builtin(__builtin_amdgcn_tanhf)
  return __builtin_amdgcn_tanhf(x);
#else
  return tanhf(x);
#endif
}

// ---- CDNA5 async global->LDS copy (16B), ASYNCcnt-tracked; sync fallback ----
__device__ __forceinline__ void async_cp16(const unsigned short* g, unsigned short* l) {
#if __has_builtin(__builtin_amdgcn_global_load_async_to_lds_b128)
  __builtin_amdgcn_global_load_async_to_lds_b128(
      (__attribute__((address_space(1))) v4i_gcc*)g,
      (__attribute__((address_space(3))) v4i_gcc*)l, 0, 0);
#else
  *reinterpret_cast<v4u*>(l) = *reinterpret_cast<const v4u*>(g);
#endif
}
__device__ __forceinline__ void async_wait0() {
#if __has_builtin(__builtin_amdgcn_global_load_async_to_lds_b128)
# if __has_builtin(__builtin_amdgcn_s_wait_asynccnt)
  __builtin_amdgcn_s_wait_asynccnt(0);
# else
  asm volatile("s_wait_asynccnt 0x0" ::: "memory");
# endif
#endif
}

// ---------------- GEMM: C[M,N] = A[M,K] * Wt[N,K]^T  (bf16 in, WMMA f32 accum)
// EPI 0: Cf = acc
// EPI 1: Cf += acc
// EPI 2: o = S * 2*sigmoid(acc); Cf = o; Cb = bf16(o)
// EPI 3: Cb = bf16(relu(acc))
#define TM 128
#define TN 128
#define TK 32
#define LSTR 40   // halves per LDS row (80 B, keeps 16B alignment)

template <int EPI>
__global__ __launch_bounds__(256)
void gemm_bf16_wmma(const unsigned short* __restrict__ A,
                    const unsigned short* __restrict__ Wt,
                    float* __restrict__ Cf, unsigned short* __restrict__ Cb,
                    const float* __restrict__ S, int M, int N, int K) {
  __shared__ __align__(16) unsigned short lA[2][TM * LSTR];
  __shared__ __align__(16) unsigned short lB[2][TN * LSTR];

  const int tid  = threadIdx.x;
  const int lane = tid & 31;
  const int wave = tid >> 5;
  const int wm   = wave >> 2;           // 0..1  (64-row stripe)
  const int wn   = wave & 3;            // 0..3  (32-col stripe)
  const int tileM = blockIdx.y * TM;
  const int tileN = blockIdx.x * TN;
  const int half = lane >> 4;           // 0/1
  const int l15  = lane & 15;

  v8f acc[4][2];
#pragma unroll
  for (int mi = 0; mi < 4; ++mi)
#pragma unroll
    for (int ni = 0; ni < 2; ++ni) acc[mi][ni] = {};

  // tile = 128 rows x 32 halves = 512 16B chunks; 2 A-chunks + 2 B-chunks/thread
  auto prefetch = [&](int buf, int kk) {
#pragma unroll
    for (int i = 0; i < 2; ++i) {
      const int c   = tid + 256 * i;    // 0..511
      const int row = c >> 2;
      const int q   = c & 3;
      async_cp16(A  + (size_t)(tileM + row) * K + kk + q * 8,
                 &lA[buf][row * LSTR + q * 8]);
      async_cp16(Wt + (size_t)(tileN + row) * K + kk + q * 8,
                 &lB[buf][row * LSTR + q * 8]);
    }
  };

  prefetch(0, 0);
  int cur = 0;
  for (int k0 = 0; k0 < K; k0 += TK) {
    async_wait0();        // this wave's copies into lX[cur] are done
    __syncthreads();      // everyone's are done
    if (k0 + TK < K) prefetch(cur ^ 1, k0 + TK);

    // A frag: lane<16 holds K 0-7 & 16-23 of row M=l15, lane>=16 holds K 8-15 & 24-31
    v16bf aF[4], bF[2];
#pragma unroll
    for (int mi = 0; mi < 4; ++mi) {
      const unsigned short* p = &lA[cur][(wm * 64 + mi * 16 + l15) * LSTR + half * 8];
      v4u* d = reinterpret_cast<v4u*>(&aF[mi]);
      d[0] = *reinterpret_cast<const v4u*>(p);
      d[1] = *reinterpret_cast<const v4u*>(p + 16);
    }
    // B frag: lane<16 holds K 0-7 & 16-23 of column N=l15, lane>=16 holds K 8-15 & 24-31
#pragma unroll
    for (int ni = 0; ni < 2; ++ni) {
      const unsigned short* p = &lB[cur][(wn * 32 + ni * 16 + l15) * LSTR + half * 8];
      v4u* d = reinterpret_cast<v4u*>(&bF[ni]);
      d[0] = *reinterpret_cast<const v4u*>(p);
      d[1] = *reinterpret_cast<const v4u*>(p + 16);
    }

#pragma unroll
    for (int mi = 0; mi < 4; ++mi)
#pragma unroll
      for (int ni = 0; ni < 2; ++ni)
        acc[mi][ni] = __builtin_amdgcn_wmma_f32_16x16x32_bf16(
            false, aF[mi], false, bF[ni], (short)0, acc[mi][ni], false, false);
    __syncthreads();
    cur ^= 1;
  }

  // epilogue: VGPR r holds M=r (lanes 0-15) / M=r+8 (lanes 16-31); N = lane&15
#pragma unroll
  for (int mi = 0; mi < 4; ++mi)
#pragma unroll
    for (int ni = 0; ni < 2; ++ni) {
      const int bm = tileM + wm * 64 + mi * 16 + half * 8;
      const int bn = tileN + wn * 32 + ni * 16 + l15;
#pragma unroll
      for (int r = 0; r < 8; ++r) {
        const size_t off = (size_t)(bm + r) * N + bn;
        const float v = acc[mi][ni][r];
        if (EPI == 0) {
          Cf[off] = v;
        } else if (EPI == 1) {
          Cf[off] += v;
        } else if (EPI == 2) {
          const float o = S[off] * 2.0f * fast_sig(v);
          Cf[off] = o;
          Cb[off] = f2bf(o);
        } else {
          Cb[off] = f2bf(fmaxf(v, 0.0f));
        }
      }
    }
}

// ---------------- fp32 -> bf16 bulk convert (weights, once per launch) -------
__global__ __launch_bounds__(256)
void cvt_bf16_kernel(const float* __restrict__ X, unsigned short* __restrict__ Y) {
  const size_t t = (size_t)blockIdx.x * 256 + threadIdx.x;   // float4 index
  const float4 f = reinterpret_cast<const float4*>(X)[t];
  ushort4 u; u.x = f2bf(f.x); u.y = f2bf(f.y); u.z = f2bf(f.z); u.w = f2bf(f.w);
  reinterpret_cast<ushort4*>(Y)[t] = u;
}

// ---------------- LayerNorm over rows of DIM (writes f32 + bf16) -------------
__global__ __launch_bounds__(256)
void layernorm_kernel(const float* __restrict__ X, const float* __restrict__ g,
                      const float* __restrict__ b, float* __restrict__ Y,
                      unsigned short* __restrict__ Yb) {
  __shared__ float red[256];
  const int row = blockIdx.x;
  const float* x = X + (size_t)row * DIM;
  float v[8], s = 0.0f;
#pragma unroll
  for (int i = 0; i < 8; ++i) { v[i] = x[threadIdx.x + 256 * i]; s += v[i]; }
  red[threadIdx.x] = s; __syncthreads();
  for (int o = 128; o > 0; o >>= 1) {
    if (threadIdx.x < o) red[threadIdx.x] += red[threadIdx.x + o];
    __syncthreads();
  }
  const float mu = red[0] * (1.0f / DIM);
  __syncthreads();
  s = 0.0f;
#pragma unroll
  for (int i = 0; i < 8; ++i) { float d = v[i] - mu; s += d * d; }
  red[threadIdx.x] = s; __syncthreads();
  for (int o = 128; o > 0; o >>= 1) {
    if (threadIdx.x < o) red[threadIdx.x] += red[threadIdx.x + o];
    __syncthreads();
  }
  const float rstd = rsqrtf(red[0] * (1.0f / DIM) + 1e-6f);
#pragma unroll
  for (int i = 0; i < 8; ++i) {
    const int c = threadIdx.x + 256 * i;
    const float y = (v[i] - mu) * rstd * g[c] + b[c];
    Y[(size_t)row * DIM + c]  = y;
    Yb[(size_t)row * DIM + c] = f2bf(y);
  }
}

// ---------------- LSTM gates (writes c,h f32 to d_out; h bf16 to ws) ---------
__global__ __launch_bounds__(256)
void gate_kernel(const float* __restrict__ lin, const float* __restrict__ bias,
                 const float* __restrict__ c_prev,
                 float* __restrict__ c_out, float* __restrict__ h_out,
                 unsigned short* __restrict__ h_bf) {
  const size_t t = (size_t)blockIdx.x * 256 + threadIdx.x;   // over B*D
  const int bi = (int)(t / DIM);
  const int n  = (int)(t % DIM);
  const float* lr = lin + (size_t)bi * 4 * DIM;
  const float ig = fast_sig(lr[n]           + bias[n]);
  const float fg = fast_sig(lr[DIM + n]     + bias[DIM + n]);
  const float og = fast_sig(lr[2 * DIM + n] + bias[2 * DIM + n]);
  const float ct = fast_tanh(lr[3 * DIM + n] + bias[3 * DIM + n]);
  const float c = fg * c_prev[t] + ig * ct;
  const float h = og * fast_tanh(c);
  c_out[t] = c;
  h_out[t] = h;
  h_bf[t]  = f2bf(h);
}

// ---------------- Embedding gather --------------------------------------------
__global__ __launch_bounds__(256)
void gather_kernel(const int* __restrict__ x, const float* __restrict__ embed,
                   float* __restrict__ out) {
  const size_t t = (size_t)blockIdx.x * 256 + threadIdx.x;
  const int bi = (int)(t / DIM);
  const int d  = (int)(t % DIM);
  out[t] = embed[(size_t)x[bi] * DIM + d];
}

// ---------------- Orchestration ----------------------------------------------
extern "C" void kernel_launch(void* const* d_in, const int* in_sizes, int n_in,
                              void* d_out, int out_size, void* d_ws, size_t ws_size,
                              hipStream_t stream) {
  const int*   x      = (const int*)  d_in[0];
  const float* c_prev = (const float*)d_in[1];
  const float* h_prev = (const float*)d_in[2];
  const float* embed  = (const float*)d_in[3];
  const float* Wm     = (const float*)d_in[4];
  const float* W      = (const float*)d_in[5];
  const float* U      = (const float*)d_in[6];
  const float* bias   = (const float*)d_in[7];
  const float* hproj  = (const float*)d_in[8];
  const float* f1     = (const float*)d_in[9];
  const float* f2     = (const float*)d_in[10];
  const float* ln1g   = (const float*)d_in[11];
  const float* ln1b   = (const float*)d_in[12];
  const float* ln2g   = (const float*)d_in[13];
  const float* ln2b   = (const float*)d_in[14];

  const size_t BD = (size_t)BATCH * DIM;
  const size_t DD = (size_t)DIM * DIM;
  float* c_out = (float*)d_out;
  float* h_out = c_out + (size_t)L_LAYERS * BD;
  float* y_out = h_out + (size_t)L_LAYERS * BD;

  // ---- workspace carve-up ----
  float* fp = (float*)d_ws;
  float* cur  = fp; fp += BD;           // layer input (f32)
  float* nxt  = fp; fp += BD;           // layer output (f32)
  float* xn_f = fp; fp += BD;           // LN output (f32, S for r=1)
  float* hm_f = fp; fp += BD;           // modulated h (f32, S for r=2)
  float* xm_f = fp; fp += BD;           // modulated x (f32)
  float* big_f = fp; fp += 4 * BD;      // lin (f32)
  unsigned short* up = (unsigned short*)fp;
  unsigned short* xn_b  = up; up += BD;
  unsigned short* hm_b  = up; up += BD;
  unsigned short* xm_b  = up; up += BD;
  unsigned short* h_b   = up; up += BD;
  unsigned short* big_b = up; up += 4 * BD;
  // bf16 weights (converted once per launch)
  unsigned short* Wm_b = up; up += (size_t)L_LAYERS * 3 * DD;
  unsigned short* W_b  = up; up += (size_t)L_LAYERS * 4 * DD;
  unsigned short* U_b  = up; up += (size_t)L_LAYERS * 4 * DD;
  unsigned short* hp_b = up; up += (size_t)L_LAYERS * DD;
  unsigned short* f1_b = up; up += (size_t)L_LAYERS * 4 * DD;
  unsigned short* f2_b = up; up += (size_t)L_LAYERS * 4 * DD;

  const dim3 blk(256);
  const dim3 gD (DIM / TN,     BATCH / TM);   // N = 2048
  const dim3 g4D(4 * DIM / TN, BATCH / TM);   // N = 8192
  const int  nBD = (int)(BD / 256);

  // convert all weights fp32 -> bf16
  cvt_bf16_kernel<<<(int)(L_LAYERS * 3 * DD / 1024), blk, 0, stream>>>(Wm, Wm_b);
  cvt_bf16_kernel<<<(int)(L_LAYERS * 4 * DD / 1024), blk, 0, stream>>>(W,  W_b);
  cvt_bf16_kernel<<<(int)(L_LAYERS * 4 * DD / 1024), blk, 0, stream>>>(U,  U_b);
  cvt_bf16_kernel<<<(int)(L_LAYERS * DD / 1024),     blk, 0, stream>>>(hproj, hp_b);
  cvt_bf16_kernel<<<(int)(L_LAYERS * 4 * DD / 1024), blk, 0, stream>>>(f1, f1_b);
  cvt_bf16_kernel<<<(int)(L_LAYERS * 4 * DD / 1024), blk, 0, stream>>>(f2, f2_b);

  gather_kernel<<<nBD, blk, 0, stream>>>(x, embed, cur);

  for (int m = 0; m < L_LAYERS; ++m) {
    const unsigned short* Wm_m = Wm_b + (size_t)m * 3 * DD;
    const unsigned short* W_m  = W_b  + (size_t)m * 4 * DD;
    const unsigned short* U_m  = U_b  + (size_t)m * 4 * DD;
    const unsigned short* hp_m = hp_b + (size_t)m * DD;
    const unsigned short* f1_m = f1_b + (size_t)m * 4 * DD;
    const unsigned short* f2_m = f2_b + (size_t)m * 4 * DD;
    const float* b_m   = bias   + (size_t)m * 4 * DIM;
    const float* cp_m  = c_prev + (size_t)m * BD;
    const float* hpr_m = h_prev + (size_t)m * BD;
    float* c_m = c_out + (size_t)m * BD;
    float* h_m = h_out + (size_t)m * BD;
    float* dst = (m == L_LAYERS - 1) ? y_out : nxt;

    // x_norm = LN1(x)
    layernorm_kernel<<<BATCH, blk, 0, stream>>>(cur, ln1g + m * DIM, ln1b + m * DIM, xn_f, xn_b);
    // r=0: hm = h_prev * 2*sigmoid(xn @ Wm0^T)
    gemm_bf16_wmma<2><<<gD, blk, 0, stream>>>(xn_b, Wm_m,          hm_f, hm_b, hpr_m, BATCH, DIM, DIM);
    // r=1: xm = xn * 2*sigmoid(hm @ Wm1^T)
    gemm_bf16_wmma<2><<<gD, blk, 0, stream>>>(hm_b, Wm_m + DD,     xm_f, xm_b, xn_f,  BATCH, DIM, DIM);
    // r=2: hm = hm * 2*sigmoid(xm @ Wm2^T)
    gemm_bf16_wmma<2><<<gD, blk, 0, stream>>>(xm_b, Wm_m + 2 * DD, hm_f, hm_b, hm_f,  BATCH, DIM, DIM);
    // lin = xm @ W^T + hm @ U^T
    gemm_bf16_wmma<0><<<g4D, blk, 0, stream>>>(xm_b, W_m, big_f, nullptr, nullptr, BATCH, 4 * DIM, DIM);
    gemm_bf16_wmma<1><<<g4D, blk, 0, stream>>>(hm_b, U_m, big_f, nullptr, nullptr, BATCH, 4 * DIM, DIM);
    // gates -> c_next, h_next
    gate_kernel<<<nBD, blk, 0, stream>>>(big_f, b_m, cp_m, c_m, h_m, h_b);
    // FFN: relu(LN2(x) @ f1^T) @ f2^T + h_next @ hproj^T
    layernorm_kernel<<<BATCH, blk, 0, stream>>>(cur, ln2g + m * DIM, ln2b + m * DIM, xn_f, xn_b);
    gemm_bf16_wmma<3><<<g4D, blk, 0, stream>>>(xn_b,  f1_m, nullptr, big_b, nullptr, BATCH, 4 * DIM, DIM);
    gemm_bf16_wmma<0><<<gD,  blk, 0, stream>>>(big_b, f2_m, dst, nullptr, nullptr, BATCH, DIM, 4 * DIM);
    gemm_bf16_wmma<1><<<gD,  blk, 0, stream>>>(h_b,   hp_m, dst, nullptr, nullptr, BATCH, DIM, DIM);

    float* t2 = cur; cur = nxt; nxt = t2;
  }
}
